// Attention_5781025980625
// MI455X (gfx1250) — matrix operand order repaired and checked by reading, not compiled, not run on hardware
//
#include <hip/hip_runtime.h>

#define B_   64
#define S_   1024
#define E_   1024   // 2*ENC_HID
#define K_   1024   // DEC_HID
#define NEG_ (-1e10f)

typedef _Float16 v4h  __attribute__((ext_vector_type(4)));
typedef _Float16 v8h  __attribute__((ext_vector_type(8)));
typedef _Float16 v16h __attribute__((ext_vector_type(16)));
typedef float    v8f  __attribute__((ext_vector_type(8)));

// ---------------------------------------------------------------------------
// Kernel 0a: hW[b,k] = hidden[b,:] @ Wh[:,k] + b_attn[k]   (full f32)
// ---------------------------------------------------------------------------
__global__ __launch_bounds__(256) void hw_kernel(const float* __restrict__ hidden,
                                                 const float* __restrict__ W,
                                                 const float* __restrict__ bias,
                                                 float* __restrict__ hW) {
    __shared__ float hrow[K_];
    const int b   = blockIdx.x;
    const int tid = threadIdx.x;
    for (int i = tid; i < K_; i += 256) hrow[i] = hidden[(size_t)b * K_ + i];
    __syncthreads();
    float acc[4] = {0.f, 0.f, 0.f, 0.f};
    for (int d = 0; d < K_; ++d) {
        const float h = hrow[d];
        const float* wr = W + (size_t)d * K_;
#pragma unroll
        for (int kk = 0; kk < 4; ++kk)
            acc[kk] += h * wr[kk * 256 + tid];
    }
#pragma unroll
    for (int kk = 0; kk < 4; ++kk) {
        const int k = kk * 256 + tid;
        hW[(size_t)b * K_ + k] = acc[kk] + bias[k];
    }
}

// ---------------------------------------------------------------------------
// Kernel 0b: WeT[n][e] = (f16) We[e][n]  — one-time 4MB->2MB transpose+convert
// ---------------------------------------------------------------------------
__global__ __launch_bounds__(256) void wet_kernel(const float* __restrict__ W,
                                                  _Float16* __restrict__ WeT) {
    __shared__ float tile[64][65];                 // +1 pad: no bank conflicts
    const int tid = threadIdx.x;
    const int e0 = blockIdx.x * 64, n0 = blockIdx.y * 64;
    const float* We = W + (size_t)K_ * K_;         // rows 1024..2047 of W_attn
    for (int i = tid; i < 64 * 64; i += 256) {
        const int r = i >> 6, c = i & 63;          // coalesced read along n
        tile[r][c] = We[(size_t)(e0 + r) * K_ + n0 + c];
    }
    __syncthreads();
    for (int i = tid; i < 64 * 64; i += 256) {
        const int n = i >> 6, e = i & 63;          // coalesced write along e
        WeT[(size_t)(n0 + n) * E_ + e0 + e] = (_Float16)tile[e][n];
    }
}

// ---------------------------------------------------------------------------
// Kernel 1: scores[b,s] = sum_k tanh(enc[b,s,:]@We[:,k] + hW[b,k]) * v[k]
// ---------------------------------------------------------------------------
__global__ __launch_bounds__(256) void scores_kernel(const float* __restrict__ enc,
                                                     const _Float16* __restrict__ WeT,
                                                     const float* __restrict__ hW,
                                                     const float* __restrict__ v,
                                                     float* __restrict__ scores) {
    __shared__ _Float16 As[64 * 128];      // [s_local][e]  16 KB
    __shared__ _Float16 BsT[128 * 128];    // [n_local][e]  32 KB
    __shared__ float    attw[8][64];       // per-wave partials, 2 KB

    const int tid  = threadIdx.x;
    const int wave = tid >> 5;
    const int lane = tid & 31;
    const int nl   = lane & 15;
    const int hi   = lane >> 4;
    const int b    = blockIdx.y;
    const int s0   = blockIdx.x * 64;

    for (int i = tid; i < 8 * 64; i += 256) ((float*)attw)[i] = 0.f;

    const float* encB = enc + ((size_t)b * S_ + s0) * E_;
    const unsigned int bst_base = (unsigned int)(uintptr_t)&BsT[0]; // LDS offset

    for (int n0 = 0; n0 < K_; n0 += 128) {
        v8f acc[4] = {};

        for (int e0 = 0; e0 < E_; e0 += 128) {
            __syncthreads();                   // prior tile reads done before overwrite

            // --- B tile: async DMA f16 WeT rows straight into LDS (no VGPRs) ---
#pragma unroll
            for (int j = 0; j < 8; ++j) {
                const int c    = tid + 256 * j;        // 16B chunk id, 0..2047
                const int nrow = c >> 4;               // 0..127
                const int col  = c & 15;               // 16B unit within row
                const unsigned int lds_off = bst_base + (unsigned int)(nrow * 256 + col * 16);
                const unsigned long long ga = (unsigned long long)(uintptr_t)
                    (WeT + ((size_t)(n0 + nrow) * E_ + e0 + col * 8));
                asm volatile("global_load_async_to_lds_b128 %0, %1, off"
                             :: "v"(lds_off), "v"(ga) : "memory");
            }

            // --- A tile: 64 x 128 f32 -> f16, clause loads then packed stores ---
            float4 av[8];
#pragma unroll
            for (int j = 0; j < 8; ++j) {
                const int f   = (tid + 256 * j) * 4;
                const int row = f >> 7, col = f & 127;
                av[j] = *(const float4*)(encB + (size_t)row * E_ + e0 + col);
            }
#pragma unroll
            for (int j = 0; j < 8; ++j) {
                const int f   = (tid + 256 * j) * 4;
                const int row = f >> 7, col = f & 127;
                v4h h;
                h[0] = (_Float16)av[j].x; h[1] = (_Float16)av[j].y;
                h[2] = (_Float16)av[j].z; h[3] = (_Float16)av[j].w;
                *(v4h*)&As[row * 128 + col] = h;       // ds_store_b64
            }

            asm volatile("s_wait_asynccnt 0x0" ::: "memory");
            __syncthreads();

            // --- 4 ek-steps; all fragment loads hoisted ahead of the WMMAs so
            //     the compiler can clause the ds_loads, wait once, and issue
            //     4 back-to-back v_wmma per step (and overlap steps). ---
#pragma unroll
            for (int ek = 0; ek < 128; ek += 32) {
                const v16h bfrag = __builtin_shufflevector(
                    *(const v8h*)&BsT[(wave * 16 + nl) * 128 + ek + hi * 8],
                    *(const v8h*)&BsT[(wave * 16 + nl) * 128 + ek + 16 + hi * 8],
                    0, 1, 2, 3, 4, 5, 6, 7, 8, 9, 10, 11, 12, 13, 14, 15);
                v16h afrag[4];
#pragma unroll
                for (int mt = 0; mt < 4; ++mt) {
                    const _Float16* arow = &As[(mt * 16 + nl) * 128 + ek];
                    const v8h alo = *(const v8h*)(arow + hi * 8);
                    const v8h ahi = *(const v8h*)(arow + 16 + hi * 8);
                    afrag[mt] = __builtin_shufflevector(
                        alo, ahi, 0, 1, 2, 3, 4, 5, 6, 7, 8, 9, 10, 11, 12, 13, 14, 15);
                }
#pragma unroll
                for (int mt = 0; mt < 4; ++mt) {
                    acc[mt] = __builtin_amdgcn_wmma_f32_16x16x32_f16(
                        false, afrag[mt], false, bfrag, (short)0, acc[mt], false, false);
                }
            }
        }

        // --- fused epilogue: tanh(pre)*v[n], deterministic shuffle reduction ---
        const int   n   = n0 + wave * 16 + nl;
        const float hwv = hW[(size_t)b * K_ + n];
        const float vn  = v[n];
#pragma unroll
        for (int mt = 0; mt < 4; ++mt) {
#pragma unroll
            for (int r = 0; r < 8; ++r) {
                float c = tanhf(acc[mt][r] + hwv) * vn;
                c += __shfl_xor(c, 1);
                c += __shfl_xor(c, 2);
                c += __shfl_xor(c, 4);
                c += __shfl_xor(c, 8);
                if (nl == 0)
                    attw[wave][mt * 16 + hi * 8 + r] += c;
            }
        }
    }
    __syncthreads();
    if (tid < 64) {
        float s = 0.f;
#pragma unroll
        for (int w = 0; w < 8; ++w) s += attw[w][tid];
        scores[(size_t)b * S_ + s0 + tid] = s;
    }
}

// ---------------------------------------------------------------------------
// Kernel 2: masked softmax over s, in place on d_out (one block per row)
// ---------------------------------------------------------------------------
__global__ __launch_bounds__(256) void softmax_kernel(const int* __restrict__ mask,
                                                      float* __restrict__ out) {
    __shared__ float red[256];
    const int b = blockIdx.x, tid = threadIdx.x;
    float x[4];
#pragma unroll
    for (int j = 0; j < 4; ++j) {
        const int s = tid + 256 * j;
        const float val = out[(size_t)b * S_ + s];
        x[j] = (mask[(size_t)b * S_ + s] == 0) ? NEG_ : val;
    }
    float m = fmaxf(fmaxf(x[0], x[1]), fmaxf(x[2], x[3]));
    red[tid] = m; __syncthreads();
    for (int off = 128; off > 0; off >>= 1) {
        if (tid < off) red[tid] = fmaxf(red[tid], red[tid + off]);
        __syncthreads();
    }
    m = red[0]; __syncthreads();
    float e[4]; float sum = 0.f;
#pragma unroll
    for (int j = 0; j < 4; ++j) { e[j] = expf(x[j] - m); sum += e[j]; }
    red[tid] = sum; __syncthreads();
    for (int off = 128; off > 0; off >>= 1) {
        if (tid < off) red[tid] += red[tid + off];
        __syncthreads();
    }
    const float inv = 1.f / red[0];
#pragma unroll
    for (int j = 0; j < 4; ++j)
        out[(size_t)b * S_ + tid + 256 * j] = e[j] * inv;
}

// ---------------------------------------------------------------------------
extern "C" void kernel_launch(void* const* d_in, const int* in_sizes, int n_in,
                              void* d_out, int out_size, void* d_ws, size_t ws_size,
                              hipStream_t stream) {
    const float* hidden = (const float*)d_in[0];   // (64,1024)
    const float* enc    = (const float*)d_in[1];   // (64,1024,1024)
    const int*   mask   = (const int*)d_in[2];     // (64,1024)
    const float* W      = (const float*)d_in[3];   // (2048,1024)
    const float* bias   = (const float*)d_in[4];   // (1024,)
    const float* v      = (const float*)d_in[5];   // (1024,)
    float* out = (float*)d_out;                    // (64,1024)

    float*    hW  = (float*)d_ws;                          // 256 KB
    _Float16* WeT = (_Float16*)((char*)d_ws + 256 * 1024); // 2 MB

    hw_kernel<<<B_, 256, 0, stream>>>(hidden, W, bias, hW);
    wet_kernel<<<dim3(E_ / 64, K_ / 64), 256, 0, stream>>>(W, WeT);
    scores_kernel<<<dim3(S_ / 64, B_), 256, 0, stream>>>(enc, WeT, hW, v, out);
    softmax_kernel<<<B_, 256, 0, stream>>>(mask, out);
}
